// RGNNPredictor_31542239822494
// MI455X (gfx1250) — compile-verified, matrix operand order checked by reading, not run on hardware
//
#include <hip/hip_runtime.h>
#include <hip/hip_bf16.h>
#include <math.h>

// ---------------------------------------------------------------------------
// Sizes (fixed by the reference problem)
// ---------------------------------------------------------------------------
#define HH      97          // hidden size
#define HP      112         // hidden padded to 7*16
#define KP      128         // K padded to 4*32 for wmma k-steps
#define INF_    9           // input features
#define NN      100000      // nodes
#define MP      100096      // nodes padded to multiple of 128 (782*128)
#define NE      800000      // edges
#define NG      4096        // graphs (multiple of 128)
#define NL      7           // layers
#define NSTEP   7           // readout steps
#define BN_EPS  1e-5f
#define WSTRIDE 48          // halfs per (k-block, column) fragment slot (32 used)

typedef __attribute__((ext_vector_type(16))) __bf16 v16bf;
typedef __attribute__((ext_vector_type(8)))  float  v8f;

// ---------------------------------------------------------------------------
// Generic bf16-WMMA GEMM:  C[M, Ntot](f32,ldc) = A[M,*](f32,lda) @ Wp + bias
//   Wp is packed in *fragment order*: for k-block ks (32 k's) and column n,
//   48 halfs at ((ks*ldw + n)*48): pos 0..15 = lanes 0-15 order
//   (k0+0..7, k0+16..23), pos 16..31 = lanes 16-31 order (k0+8..15, k0+24..31).
//   grid = (M/128, Ntot/112), block = 256 (8 waves); each wave: 16 rows x 112.
// WMMA C/D v8f: elem i -> row = 8*(lane>>4)+i, col = (lane&15).
// ---------------------------------------------------------------------------
__global__ __launch_bounds__(256)
void gemm_bf16(const float* __restrict__ A, int lda,
               const __bf16* __restrict__ W, int ldw /*padded col count*/,
               const float* __restrict__ bias,
               float* __restrict__ C, int ldc, int Ntot)
{
    __shared__ __align__(32) __bf16 Wlds[4 * HP * WSTRIDE];   // 43,008 B
    const int n0 = blockIdx.y * HP;
    {
        const int gran = HP * WSTRIDE / 8;        // uint4 granules per k-block chunk
        for (int i = threadIdx.x; i < 4 * gran; i += 256) {
            int ks = i / gran, r = i % gran;
            ((uint4*)Wlds)[(size_t)ks * gran + r] =
                ((const uint4*)(W + ((size_t)ks * ldw + n0) * WSTRIDE))[r];
        }
    }
    __syncthreads();

    const int lane = threadIdx.x & 31;
    const int wave = threadIdx.x >> 5;
    const int row0 = blockIdx.x * 128 + wave * 16;
    const int rA   = row0 + (lane & 15);
    const int half = lane >> 4;

    v8f acc[7];
#pragma unroll
    for (int t = 0; t < 7; ++t)
#pragma unroll
        for (int i = 0; i < 8; ++i) acc[t][i] = 0.0f;

#pragma unroll
    for (int ks = 0; ks < 4; ++ks) {
        const int kb = ks * 32 + half * 8;
        // ---- A fragment: rows striped over lanes, 2x8 consecutive k's -----
        const float* ap = A + (size_t)rA * lda + kb;
        float4 a0 = *(const float4*)(ap);
        float4 a1 = *(const float4*)(ap + 4);
        float4 a2, a3;
        if (ks < 3) {                // kb+16..kb+23 < 112 for ks 0..2
            a2 = *(const float4*)(ap + 16);
            a3 = *(const float4*)(ap + 20);
        } else {                     // ks==3: k >= 112 is zero padding
            a2 = make_float4(0.f, 0.f, 0.f, 0.f);
            a3 = make_float4(0.f, 0.f, 0.f, 0.f);
        }
        v16bf a;
        a[0]  = (__bf16)a0.x; a[1]  = (__bf16)a0.y; a[2]  = (__bf16)a0.z; a[3]  = (__bf16)a0.w;
        a[4]  = (__bf16)a1.x; a[5]  = (__bf16)a1.y; a[6]  = (__bf16)a1.z; a[7]  = (__bf16)a1.w;
        a[8]  = (__bf16)a2.x; a[9]  = (__bf16)a2.y; a[10] = (__bf16)a2.z; a[11] = (__bf16)a2.w;
        a[12] = (__bf16)a3.x; a[13] = (__bf16)a3.y; a[14] = (__bf16)a3.z; a[15] = (__bf16)a3.w;

        // ---- B fragments: batch all 7 contiguous 32B LDS reads, then ------
        // ---- issue the 7 independent WMMAs back-to-back -------------------
        v16bf bf[7];
#pragma unroll
        for (int t = 0; t < 7; ++t) {
            const __bf16* bp = Wlds +
                ((size_t)(ks * HP + t * 16 + (lane & 15)) * WSTRIDE + half * 16);
            bf[t] = *(const v16bf*)bp;
        }
#pragma unroll
        for (int t = 0; t < 7; ++t)
            acc[t] = __builtin_amdgcn_wmma_f32_16x16x32_bf16(
                false, a, false, bf[t], (short)0, acc[t], false, false);
    }

    const int rowC = row0 + half * 8;
#pragma unroll
    for (int t = 0; t < 7; ++t) {
        const int col = n0 + t * 16 + (lane & 15);
        if (col < Ntot) {
            const float bv = bias ? bias[col] : 0.0f;
#pragma unroll
            for (int i = 0; i < 8; ++i)
                C[(size_t)(rowC + i) * ldc + col] = acc[t][i] + bv;
        }
    }
}

// ---------------------------------------------------------------------------
// Pack f32 weights into the fragment-ordered bf16 layout described above.
// grid = KP blocks (k), HP threads (n) per 112-column destination window.
// ---------------------------------------------------------------------------
__global__ void pack_w(const float* __restrict__ W, int K, int srcLd, int srcCol0,
                       int N, __bf16* __restrict__ dst, int ldd, int dstCol0)
{
    int k = blockIdx.x, n = threadIdx.x;
    float v = (k < K && n < N) ? W[(size_t)k * srcLd + srcCol0 + n] : 0.0f;
    int ks = k >> 5, kr = k & 31;
    int pos = (kr < 8) ? kr : (kr < 16) ? kr + 8 : (kr < 24) ? kr - 8 : kr;
    dst[((size_t)ks * ldd + dstCol0 + n) * WSTRIDE + pos] = (__bf16)v;
}

__global__ void pack_bias(const float* __restrict__ b, int srcCol0, int N,
                          float* __restrict__ dst, int dstCol0)
{
    int n = threadIdx.x;                           // 112 threads
    dst[dstCol0 + n] = (n < N) ? b[srcCol0 + n] : 0.0f;
}

// ---------------------------------------------------------------------------
// lin1: x[NN,9] @ W[9,97] + b, leaky_relu(0.01), write padded [MP,112]
// ---------------------------------------------------------------------------
__global__ void lin1_kernel(const float* __restrict__ x, const float* __restrict__ W,
                            const float* __restrict__ b, float* __restrict__ out)
{
    int row = blockIdx.x, col = threadIdx.x;       // grid MP x 128
    if (col >= HP) return;
    float v = 0.0f;
    if (row < NN && col < HH) {
#pragma unroll
        for (int j = 0; j < INF_; ++j)
            v += x[(size_t)row * INF_ + j] * W[j * HH + col];
        v += b[col];
        v = v > 0.0f ? v : 0.01f * v;
    }
    out[(size_t)row * HP + col] = v;
}

// ---------------------------------------------------------------------------
// Edge messages: S holds [k|q|v|s+agg] column blocks with stride 672.
//   agg[dst] += sigmoid(k[dst]+q[src]) * v[src]
// ---------------------------------------------------------------------------
__global__ void edge_kernel(float* __restrict__ S, const int* __restrict__ ei)
{
    int e = blockIdx.x, c = threadIdx.x;           // grid NE x 128
    if (c >= HH) return;
    int s = ei[e];
    int d = ei[NE + e];
    float k = S[(size_t)d * 672 + 0   + c];
    float q = S[(size_t)s * 672 + 112 + c];
    float v = S[(size_t)s * 672 + 224 + c];
    float g = 1.0f / (1.0f + expf(-(k + q)));
    atomicAdd(&S[(size_t)d * 672 + 336 + c], g * v);
}

// ---------------------------------------------------------------------------
// h = elu(agg); optional eval-mode BatchNorm.  agg lives in S cols [336,448).
// ---------------------------------------------------------------------------
__global__ void h_elu_bn(const float* __restrict__ S, float* __restrict__ H,
                         const float* __restrict__ gma, const float* __restrict__ bta,
                         const float* __restrict__ mu,  const float* __restrict__ var)
{
    int row = blockIdx.x, col = threadIdx.x;       // grid MP x 128
    if (col >= HP) return;
    float v = 0.0f;
    if (row < NN && col < HH) {
        v = S[(size_t)row * 672 + 336 + col];
        v = v > 0.0f ? v : (expf(v) - 1.0f);
        if (gma) {
            float sc = gma[col] * rsqrtf(var[col] + BN_EPS);
            v = (v - mu[col]) * sc + bta[col];
        }
    }
    H[(size_t)row * HP + col] = v;
}

// ---------------------------------------------------------------------------
// GRU combine: gi in S cols [0,336), gh in S cols [336,672); state in X.
//   X = relu((1-z)*n + z*X)
// ---------------------------------------------------------------------------
__global__ void gru_combine(const float* __restrict__ S, float* __restrict__ X, int nvalid)
{
    int row = blockIdx.x, col = threadIdx.x;       // grid rows x 128
    if (col >= HP) return;
    float v = 0.0f;
    if (row < nvalid && col < HH) {
        size_t base = (size_t)row * 672;
        float ir = S[base + col],       iz = S[base + 112 + col], in_ = S[base + 224 + col];
        float hr = S[base + 336 + col], hz = S[base + 448 + col], hn  = S[base + 560 + col];
        float r = 1.0f / (1.0f + expf(-(ir + hr)));
        float z = 1.0f / (1.0f + expf(-(iz + hz)));
        float n = tanhf(in_ + r * hn);
        float h = X[(size_t)row * HP + col];
        v = fmaxf((1.0f - z) * n + z * h, 0.0f);
    }
    X[(size_t)row * HP + col] = v;
}

// ---------------------------------------------------------------------------
// Readout kernels
// ---------------------------------------------------------------------------
__global__ void zero2d(float* __restrict__ p)
{
    int g = blockIdx.x, c = threadIdx.x;
    if (c < HP) p[(size_t)g * HP + c] = 0.0f;
}

__global__ void pool_add(const float* __restrict__ X, const int* __restrict__ batch,
                         float* __restrict__ out)
{
    int n = blockIdx.x, c = threadIdx.x;           // grid NN x 128
    if (c >= HH) return;
    atomicAdd(&out[(size_t)batch[n] * HP + c], X[(size_t)n * HP + c]);
}

__global__ void relu2d(float* __restrict__ p)
{
    int g = blockIdx.x, c = threadIdx.x;
    if (c < HP) p[(size_t)g * HP + c] = fmaxf(p[(size_t)g * HP + c], 0.0f);
}

__global__ void dot_rows(const float* __restrict__ M, const float* __restrict__ v,
                         float* __restrict__ out, int rows)
{
    int r = blockIdx.x * blockDim.x + threadIdx.x;
    if (r >= rows) return;
    float s = 0.0f;
    for (int c = 0; c < HH; ++c) s += M[(size_t)r * HP + c] * v[c];
    out[r] = s;
}

__device__ __forceinline__ unsigned enc_f32(float f) {
    unsigned u = __float_as_uint(f);
    return (u & 0x80000000u) ? ~u : (u | 0x80000000u);
}
__device__ __forceinline__ float dec_f32(unsigned u) {
    u = (u & 0x80000000u) ? (u & 0x7FFFFFFFu) : ~u;
    return __uint_as_float(u);
}

__global__ void step_init(float* __restrict__ hg, unsigned* __restrict__ menc,
                          float* __restrict__ ssum)
{
    int g = blockIdx.x, c = threadIdx.x;           // grid NG x 128
    if (c < HP) hg[(size_t)g * HP + c] = 0.0f;
    if (c == 0) { menc[g] = 0x00800000u /* enc(-FLT_MAX) */; ssum[g] = 0.0f; }
}

__global__ void alpha_kernel(const float* __restrict__ a_s, const float* __restrict__ a_d,
                             const int* __restrict__ batch, float* __restrict__ alpha,
                             unsigned* __restrict__ menc)
{
    int n = blockIdx.x * blockDim.x + threadIdx.x;
    if (n >= NN) return;
    int b = batch[n];
    float a = a_s[n] + a_d[b];
    a = a > 0.0f ? a : 0.01f * a;
    alpha[n] = a;
    atomicMax(&menc[b], enc_f32(a));
}

__global__ void esum_kernel(const float* __restrict__ alpha, const unsigned* __restrict__ menc,
                            const int* __restrict__ batch, float* __restrict__ evec,
                            float* __restrict__ ssum)
{
    int n = blockIdx.x * blockDim.x + threadIdx.x;
    if (n >= NN) return;
    int b = batch[n];
    float e = expf(alpha[n] - dec_f32(menc[b]));
    evec[n] = e;
    atomicAdd(&ssum[b], e);
}

__global__ void wsum_kernel(const float* __restrict__ evec, const float* __restrict__ ssum,
                            const int* __restrict__ batch, const float* __restrict__ xs,
                            float* __restrict__ hg)
{
    int n = blockIdx.x, c = threadIdx.x;           // grid NN x 128
    if (c >= HH) return;
    int b = batch[n];
    float w = evec[n] / ssum[b];
    atomicAdd(&hg[(size_t)b * HP + c], w * xs[(size_t)n * HP + c]);
}

__global__ void hg_post(float* __restrict__ hg, const float* __restrict__ gatb)
{
    int g = blockIdx.x, c = threadIdx.x;           // grid NG x 128
    if (c >= HP) return;
    float v = 0.0f;
    if (c < HH) {
        v = hg[(size_t)g * HP + c] + gatb[c];
        v = v > 0.0f ? v : (expf(v) - 1.0f);
    }
    hg[(size_t)g * HP + c] = v;
}

__global__ void final_kernel(const float* __restrict__ out, const float* __restrict__ W,
                             const float* __restrict__ b, float* __restrict__ y)
{
    int g = blockIdx.x * blockDim.x + threadIdx.x;
    if (g >= NG) return;
    float s = 0.0f;
    for (int c = 0; c < HH; ++c) s += out[(size_t)g * HP + c] * W[c];
    y[g] = s + b[0];
}

// ---------------------------------------------------------------------------
// Host driver — graph-capture safe (only kernel launches on `stream`).
// Workspace usage ~380 MB.
// ---------------------------------------------------------------------------
extern "C" void kernel_launch(void* const* d_in, const int* in_sizes, int n_in,
                              void* d_out, int out_size, void* d_ws, size_t ws_size,
                              hipStream_t stream)
{
    (void)in_sizes; (void)n_in; (void)out_size; (void)ws_size;

    const float* x      = (const float*)d_in[0];
    const int*   ei     = (const int*)d_in[1];
    const int*   batch  = (const int*)d_in[2];
    const float* lin1W  = (const float*)d_in[4];
    const float* lin1b  = (const float*)d_in[5];
    const float* convW[4] = {(const float*)d_in[6], (const float*)d_in[7],
                             (const float*)d_in[8], (const float*)d_in[9]};
    const float* convB[4] = {(const float*)d_in[10], (const float*)d_in[11],
                             (const float*)d_in[12], (const float*)d_in[13]};
    const float* gruWih = (const float*)d_in[14];
    const float* gruWhh = (const float*)d_in[15];
    const float* gruBih = (const float*)d_in[16];
    const float* gruBhh = (const float*)d_in[17];
    const float* bnG    = (const float*)d_in[18];
    const float* bnB    = (const float*)d_in[19];
    const float* bnM    = (const float*)d_in[20];
    const float* bnV    = (const float*)d_in[21];
    const float* gatW   = (const float*)d_in[22];
    const float* asrc   = (const float*)d_in[23];
    const float* adst   = (const float*)d_in[24];
    const float* gatb   = (const float*)d_in[25];
    const float* molWih = (const float*)d_in[26];
    const float* molWhh = (const float*)d_in[27];
    const float* molBih = (const float*)d_in[28];
    const float* molBhh = (const float*)d_in[29];
    const float* lin2W  = (const float*)d_in[30];
    const float* lin2b  = (const float*)d_in[31];
    float*       y      = (float*)d_out;

    // ---- workspace bump allocator -----------------------------------------
    char* ws = (char*)d_ws;
    size_t off = 0;
    auto carve = [&](size_t bytes) -> void* {
        void* p = ws + off;
        off = (off + bytes + 255) & ~size_t(255);
        return p;
    };
    // packed weight matrix bytes: 4 k-blocks * Npad cols * WSTRIDE halfs * 2B
    auto wbytes = [](int npad) -> size_t { return (size_t)4 * npad * WSTRIDE * 2; };

    float*    XBUF   = (float*)carve((size_t)MP * HP * 4);      // node state x
    float*    HBUF   = (float*)carve((size_t)MP * HP * 4);      // h / later xs
    float*    SBUF   = (float*)carve((size_t)MP * 672 * 4);     // kqvs / gi|gh
    float*    OUTG   = (float*)carve((size_t)NG * HP * 4);      // graph state
    float*    ODG    = (float*)carve((size_t)NG * HP * 4);      // out @ gatW
    float*    HGG    = (float*)carve((size_t)NG * HP * 4);      // attention agg
    float*    SG     = (float*)carve((size_t)NG * 672 * 4);     // mol gru gates
    float*    A_S    = (float*)carve((size_t)MP * 4);
    float*    ALPHA  = (float*)carve((size_t)MP * 4);
    float*    EVEC   = (float*)carve((size_t)MP * 4);
    float*    A_D    = (float*)carve((size_t)NG * 4);
    unsigned* MENC   = (unsigned*)carve((size_t)NG * 4);
    float*    SSUM   = (float*)carve((size_t)NG * 4);
    __bf16*   convWp = (__bf16*)carve((size_t)NL * wbytes(448));
    float*    convBp = (float*)carve((size_t)NL * 448 * 4);
    __bf16*   gWihP  = (__bf16*)carve((size_t)NL * wbytes(336));
    __bf16*   gWhhP  = (__bf16*)carve((size_t)NL * wbytes(336));
    float*    gBihP  = (float*)carve((size_t)NL * 336 * 4);
    float*    gBhhP  = (float*)carve((size_t)NL * 336 * 4);
    __bf16*   mWihP  = (__bf16*)carve(wbytes(336));
    __bf16*   mWhhP  = (__bf16*)carve(wbytes(336));
    float*    mBihP  = (float*)carve(336 * 4);
    float*    mBhhP  = (float*)carve(336 * 4);
    __bf16*   gatWp  = (__bf16*)carve(wbytes(112));

    const size_t wE448 = wbytes(448) / 2;   // elements per packed conv matrix
    const size_t wE336 = wbytes(336) / 2;

    // ---- pack weights/biases into fragment-ordered bf16 + padded f32 ------
    for (int l = 0; l < NL; ++l) {
        for (int j = 0; j < 4; ++j) {
            pack_w<<<KP, HP, 0, stream>>>(convW[j] + (size_t)l * HH * HH, HH, HH, 0, HH,
                                          convWp + (size_t)l * wE448, 448, j * HP);
            pack_bias<<<1, HP, 0, stream>>>(convB[j] + (size_t)l * HH, 0, HH,
                                            convBp + (size_t)l * 448, j * HP);
        }
        for (int c = 0; c < 3; ++c) {
            pack_w<<<KP, HP, 0, stream>>>(gruWih + (size_t)l * HH * 3 * HH, HH, 3 * HH, c * HH, HH,
                                          gWihP + (size_t)l * wE336, 336, c * HP);
            pack_w<<<KP, HP, 0, stream>>>(gruWhh + (size_t)l * HH * 3 * HH, HH, 3 * HH, c * HH, HH,
                                          gWhhP + (size_t)l * wE336, 336, c * HP);
            pack_bias<<<1, HP, 0, stream>>>(gruBih + (size_t)l * 3 * HH, c * HH, HH,
                                            gBihP + (size_t)l * 336, c * HP);
            pack_bias<<<1, HP, 0, stream>>>(gruBhh + (size_t)l * 3 * HH, c * HH, HH,
                                            gBhhP + (size_t)l * 336, c * HP);
        }
    }
    for (int c = 0; c < 3; ++c) {
        pack_w<<<KP, HP, 0, stream>>>(molWih, HH, 3 * HH, c * HH, HH, mWihP, 336, c * HP);
        pack_w<<<KP, HP, 0, stream>>>(molWhh, HH, 3 * HH, c * HH, HH, mWhhP, 336, c * HP);
        pack_bias<<<1, HP, 0, stream>>>(molBih, c * HH, HH, mBihP, c * HP);
        pack_bias<<<1, HP, 0, stream>>>(molBhh, c * HH, HH, mBhhP, c * HP);
    }
    pack_w<<<KP, HP, 0, stream>>>(gatW, HH, HH, 0, HH, gatWp, 112, 0);

    // ---- input projection --------------------------------------------------
    lin1_kernel<<<MP, 128, 0, stream>>>(x, lin1W, lin1b, XBUF);

    // ---- 7 x (ResGatedGraphConv -> [BN] -> GRU) ----------------------------
    const dim3 gBig(MP / 128, 4);   // N=448
    const dim3 gGru(MP / 128, 3);   // N=336
    for (int l = 0; l < NL; ++l) {
        gemm_bf16<<<gBig, 256, 0, stream>>>(XBUF, HP, convWp + (size_t)l * wE448, 448,
                                            convBp + (size_t)l * 448, SBUF, 672, 448);
        edge_kernel<<<NE, 128, 0, stream>>>(SBUF, ei);
        if (l == 0)
            h_elu_bn<<<MP, 128, 0, stream>>>(SBUF, HBUF, nullptr, nullptr, nullptr, nullptr);
        else
            h_elu_bn<<<MP, 128, 0, stream>>>(SBUF, HBUF,
                                             bnG + (size_t)(l - 1) * HH, bnB + (size_t)(l - 1) * HH,
                                             bnM + (size_t)(l - 1) * HH, bnV + (size_t)(l - 1) * HH);
        gemm_bf16<<<gGru, 256, 0, stream>>>(HBUF, HP, gWihP + (size_t)l * wE336, 336,
                                            gBihP + (size_t)l * 336, SBUF, 672, 336);
        gemm_bf16<<<gGru, 256, 0, stream>>>(XBUF, HP, gWhhP + (size_t)l * wE336, 336,
                                            gBhhP + (size_t)l * 336, SBUF + 336, 672, 336);
        gru_combine<<<MP, 128, 0, stream>>>(SBUF, XBUF, NN);
    }

    // ---- readout: global_add_pool + 7-step bipartite GAT + molecule GRU ----
    zero2d<<<NG, 128, 0, stream>>>(OUTG);
    pool_add<<<NN, 128, 0, stream>>>(XBUF, batch, OUTG);
    relu2d<<<NG, 128, 0, stream>>>(OUTG);

    // xs = x @ gat_W  (into HBUF, fixed over timesteps); a_s = xs @ asrc
    gemm_bf16<<<dim3(MP / 128, 1), 256, 0, stream>>>(XBUF, HP, gatWp, 112, nullptr, HBUF, HP, HP);
    dot_rows<<<(NN + 255) / 256, 256, 0, stream>>>(HBUF, asrc, A_S, NN);

    const dim3 gG1(NG / 128, 1);
    const dim3 gG3(NG / 128, 3);
    for (int s = 0; s < NSTEP; ++s) {
        gemm_bf16<<<gG1, 256, 0, stream>>>(OUTG, HP, gatWp, 112, nullptr, ODG, HP, HP);
        dot_rows<<<(NG + 255) / 256, 256, 0, stream>>>(ODG, adst, A_D, NG);
        step_init<<<NG, 128, 0, stream>>>(HGG, MENC, SSUM);
        alpha_kernel<<<(NN + 255) / 256, 256, 0, stream>>>(A_S, A_D, batch, ALPHA, MENC);
        esum_kernel<<<(NN + 255) / 256, 256, 0, stream>>>(ALPHA, MENC, batch, EVEC, SSUM);
        wsum_kernel<<<NN, 128, 0, stream>>>(EVEC, SSUM, batch, HBUF, HGG);
        hg_post<<<NG, 128, 0, stream>>>(HGG, gatb);
        gemm_bf16<<<gG3, 256, 0, stream>>>(HGG, HP, mWihP, 336, mBihP, SG, 672, 336);
        gemm_bf16<<<gG3, 256, 0, stream>>>(OUTG, HP, mWhhP, 336, mBhhP, SG + 336, 672, 336);
        gru_combine<<<NG, 128, 0, stream>>>(SG, OUTG, NG);
    }

    final_kernel<<<(NG + 255) / 256, 256, 0, stream>>>(OUTG, lin2W, lin2b, y);
}
